// LocalDecoderAddBaseline_72962904424984
// MI455X (gfx1250) — matrix operand order repaired and checked by reading, not compiled, not run on hardware
//
#include <hip/hip_runtime.h>

typedef __attribute__((ext_vector_type(16))) __bf16 v16bf;
typedef __attribute__((ext_vector_type(8)))  __bf16 v8bf;
typedef __attribute__((ext_vector_type(8)))  float  v8f;

#define NB    4
#define NPTS  131072
#define NC    128
#define NG    64
#define NG3   (NG*NG*NG)
#define NH    32
#define NEG_SLOPE 0.01f

#define WAVES_PER_BLOCK 4
#define PTS_PER_WAVE    16
#define PTS_PER_BLOCK   (WAVES_PER_BLOCK * PTS_PER_WAVE)
#define CPAD   136   // 128 + 8 bf16 pad -> 272B rows (16B aligned, bank-staggered)
#define WBPAD  48    // 32 + 16 bf16 pad -> 96B rows (16B aligned)
#define TVOX   32    // voxels per transpose tile

__device__ __forceinline__ float unnorm_coord(float g, float size) {
    float t = 2.0f * g / (size - 1.0f) - 1.0f;
    t = fminf(fmaxf(t, -2.0f), 2.0f);
    float x = (t + 1.0f) * 0.5f * (size - 1.0f);
    return fminf(fmaxf(x, 0.0f), size - 1.0f);
}

// Column swizzle within a 32-wide k-block: swap bit3 <-> bit4.
// Stored order per block: K {0..7}, {16..23}, {8..15}, {24..31} so that each
// lane-half's A/net fragment is one contiguous 32B chunk (2x ds_load_b128).
__device__ __forceinline__ int swz32(int r) {
    return (r & 7) | ((r & 8) << 1) | ((r & 16) >> 1);
}

// Load 16 consecutive bf16 (32B) from 16B-aligned LDS into a v16bf.
__device__ __forceinline__ v16bf load_frag32B(const __bf16* __restrict__ p) {
    v8bf lo = *(const v8bf*)(p);
    v8bf hi = *(const v8bf*)(p + 8);
    v16bf r;
#pragma unroll
    for (int i = 0; i < 8; ++i) { r[i] = lo[i]; r[i + 8] = hi[i]; }
    return r;
}

// ---------------------------------------------------------------------------
// Kernel 1: transpose c_plane [B][C][V] f32 -> c_vox [B][V][C] bf16 so the
// trilinear gather reads 256B of fully-used contiguous lines per corner
// instead of 128 separate cache lines (1MB channel stride).
// ---------------------------------------------------------------------------
__global__ __launch_bounds__(256) void transpose_kernel(
    const float* __restrict__ c_plane, __bf16* __restrict__ c_vox)
{
    __shared__ __align__(16) __bf16 tile[TVOX][NC + 8];

    const int vpb   = NG3 / TVOX;
    const int b     = blockIdx.x / vpb;
    const int vbase = (blockIdx.x % vpb) * TVOX;
    const float* src = c_plane + (size_t)b * NC * NG3;

    const int tv = threadIdx.x & 31;          // voxel within tile
    const int tc = threadIdx.x >> 5;          // 8 channel slices
    for (int c0 = 0; c0 < NC; c0 += 8) {      // coalesced reads (32 voxels/row)
        int c = c0 + tc;
        tile[tv][c] = (__bf16)src[(size_t)c * NG3 + vbase + tv];
    }
    __syncthreads();

    __bf16* dst = c_vox + ((size_t)b * NG3 + vbase) * NC;
    for (int k = threadIdx.x; k < TVOX * (NC / 8); k += 256) {  // coalesced 16B stores
        int v = k >> 4, part = k & 15;
        *(v8bf*)(dst + (size_t)v * NC + part * 8) = *(const v8bf*)(&tile[v][part * 8]);
    }
}

// ---------------------------------------------------------------------------
// Main fused kernel. TR=1: gather from voxel-major bf16 c_vox (dense reads);
// TR=0: fallback scalar gather straight from channel-major c_plane.
// ---------------------------------------------------------------------------
template <bool TR>
__global__ __launch_bounds__(128) void local_decoder_kernel(
    const float* __restrict__ pcl_mem,
    const float* __restrict__ c_plane,
    const __bf16* __restrict__ c_vox,
    const float* __restrict__ W_p,  const float* __restrict__ b_p,
    const float* __restrict__ W_c1, const float* __restrict__ b_c1,
    const float* __restrict__ W_c2, const float* __restrict__ b_c2,
    const float* __restrict__ W_b1, const float* __restrict__ b_b1,
    const float* __restrict__ W_b2, const float* __restrict__ b_b2,
    const float* __restrict__ W_out, const float* __restrict__ b_out,
    float* __restrict__ out)
{
    // Weights stored TRANSPOSED: row = output column n, contiguous in k.
    __shared__ __align__(16) __bf16 s_wc1T[NH * CPAD];
    __shared__ __align__(16) __bf16 s_wc2T[NH * CPAD];
    __shared__ __align__(16) __bf16 s_wb1T[NH * WBPAD];
    __shared__ __align__(16) __bf16 s_wb2T[NH * WBPAD];
    __shared__ __align__(16) __bf16 s_c[WAVES_PER_BLOCK][PTS_PER_WAVE][CPAD];
    __shared__ __align__(16) __bf16 s_net[WAVES_PER_BLOCK][PTS_PER_WAVE][WBPAD];
    __shared__ float s_pn[WAVES_PER_BLOCK][PTS_PER_WAVE][4];
    __shared__ int   s_pidx[WAVES_PER_BLOCK][PTS_PER_WAVE][8];
    __shared__ float s_pw[WAVES_PER_BLOCK][PTS_PER_WAVE][8];

    const int tid = threadIdx.x;
    // Stage weights to LDS as bf16, transposed to [n][k]
    for (int i = tid; i < NC * NH; i += blockDim.x) {
        int k = i >> 5, n = i & 31;          // i = k*32 + n (coalesced global read)
        s_wc1T[n * CPAD + (k & ~31) + swz32(k & 31)] = (__bf16)W_c1[i];
        s_wc2T[n * CPAD + (k & ~31) + swz32(k & 31)] = (__bf16)W_c2[i];
    }
    for (int i = tid; i < NH * NH; i += blockDim.x) {
        int k = i >> 5, n = i & 31;
        s_wb1T[n * WBPAD + swz32(k)] = (__bf16)W_b1[i];
        s_wb2T[n * WBPAD + swz32(k)] = (__bf16)W_b2[i];
    }

    const int blocksPerBatch = NPTS / PTS_PER_BLOCK;
    const int b    = blockIdx.x / blocksPerBatch;
    const int tb   = (blockIdx.x % blocksPerBatch) * PTS_PER_BLOCK;
    const int wave = tid >> 5;
    const int lane = tid & 31;
    const int half = lane >> 4;   // 0 or 1
    const int l15  = lane & 15;
    const int pt   = l15;                                // point within wave tile
    const int gpt  = tb + wave * PTS_PER_WAVE + pt;      // point index within batch

    // ---- per-point trilinear setup (lane L and L+16 share the same point) ----
    const size_t pbase = ((size_t)b * NPTS + gpt) * 3;
    float gx = pcl_mem[pbase + 0];
    float gy = pcl_mem[pbase + 1];
    float gz = pcl_mem[pbase + 2];

    float ix = unnorm_coord(gx, (float)NG);
    float iy = unnorm_coord(gy, (float)NG);
    float iz = unnorm_coord(gz, (float)NG);

    float x0f = floorf(ix), y0f = floorf(iy), z0f = floorf(iz);
    float wx = ix - x0f, wy = iy - y0f, wz = iz - z0f;

    int x0 = min(max((int)x0f, 0), NG - 1); int x1 = min(x0 + 1, NG - 1);
    int y0 = min(max((int)y0f, 0), NG - 1); int y1 = min(y0 + 1, NG - 1);
    int z0 = min(max((int)z0f, 0), NG - 1); int z1 = min(z0 + 1, NG - 1);

    int idx8[8]; float w8[8];
    {
        int zi[2] = { z0, z1 }; float wzv[2] = { 1.0f - wz, wz };
        int yi[2] = { y0, y1 }; float wyv[2] = { 1.0f - wy, wy };
        int xi[2] = { x0, x1 }; float wxv[2] = { 1.0f - wx, wx };
#pragma unroll
        for (int j = 0; j < 8; ++j) {
            int zz = (j >> 2) & 1, yy = (j >> 1) & 1, xx = j & 1;
            idx8[j] = (zi[zz] * NG + yi[yy]) * NG + xi[xx];
            w8[j]   = wzv[zz] * wyv[yy] * wxv[xx];
        }
    }

    if (half == 0) {  // pcl_norm = pcl_mem - trunc(pcl_mem) - 0.5
        s_pn[wave][pt][0] = gx - truncf(gx) - 0.5f;
        s_pn[wave][pt][1] = gy - truncf(gy) - 0.5f;
        s_pn[wave][pt][2] = gz - truncf(gz) - 0.5f;
        if (TR) {
#pragma unroll
            for (int j = 0; j < 8; ++j) {
                s_pidx[wave][pt][j] = idx8[j];
                s_pw[wave][pt][j]   = w8[j];
            }
        }
    }

    if (TR) {
        // ---- dense gather: 8 corners x 256B contiguous bf16 rows ----
        __syncthreads();
        const int ptq = lane >> 2;     // 8 points per pass
        const int cq  = lane & 3;      // 4 quarters of 32 channels
        const __bf16* vb = c_vox + (size_t)b * NG3 * NC;
#pragma unroll
        for (int pass = 0; pass < 2; ++pass) {
            const int p = pass * 8 + ptq;
            float accv[32];
#pragma unroll
            for (int i = 0; i < 32; ++i) accv[i] = 0.0f;
#pragma unroll
            for (int j = 0; j < 8; ++j) {
                const int   idx = s_pidx[wave][p][j];
                const float wj  = s_pw[wave][p][j];
                const __bf16* row = vb + (size_t)idx * NC + cq * 32;
#pragma unroll
                for (int q = 0; q < 4; ++q) {
                    v8bf v = *(const v8bf*)(row + q * 8);
#pragma unroll
                    for (int e = 0; e < 8; ++e)
                        accv[q * 8 + e] = fmaf((float)v[e], wj, accv[q * 8 + e]);
                }
            }
            // store one full k-block (kb = cq), swizzled, as 4x ds_store_b128
            const int gmap[4] = { 0, 16, 8, 24 };
#pragma unroll
            for (int g = 0; g < 4; ++g) {
                v8bf o;
#pragma unroll
                for (int e = 0; e < 8; ++e) o[e] = (__bf16)accv[g * 8 + e];
                *(v8bf*)(&s_c[wave][p][cq * 32 + gmap[g]]) = o;
            }
        }
    } else {
        // ---- fallback: scattered scalar gather from channel-major volume ----
        const float* cbase = c_plane + (size_t)b * NC * NG3;
        for (int cb = 0; cb < NC; cb += 2) {
            const int ch = cb + half;
            const float* p = cbase + (size_t)ch * NG3;
            if (cb + 2 < NC) {
                const float* pn = p + 2 * (size_t)NG3;
#pragma unroll
                for (int j = 0; j < 8; ++j) __builtin_prefetch(pn + idx8[j], 0, 1);
            }
            float acc = 0.0f;
#pragma unroll
            for (int j = 0; j < 8; ++j) acc = fmaf(p[idx8[j]], w8[j], acc);
            const int pos = (ch & ~31) + swz32(ch & 31);
            s_c[wave][pt][pos] = (__bf16)acc;
        }
    }
    __syncthreads();

    // ---- init net = pcl_norm @ W_p + b_p, directly in WMMA C/D layout ----
    // acc0: cols 0..15 (N = l15), acc1: cols 16..31. VGPR r <-> M = half*8 + r.
    v8f acc0, acc1;
    {
        const int n0 = l15, n1 = 16 + l15;
        float bp0 = b_p[n0], bp1 = b_p[n1];
        float wp[3][2];
#pragma unroll
        for (int d = 0; d < 3; ++d) { wp[d][0] = W_p[d * NH + n0]; wp[d][1] = W_p[d * NH + n1]; }
#pragma unroll
        for (int r = 0; r < 8; ++r) {
            int m = half * 8 + r;
            float px = s_pn[wave][m][0];
            float py = s_pn[wave][m][1];
            float pz = s_pn[wave][m][2];
            acc0[r] = bp0 + px * wp[0][0] + py * wp[1][0] + pz * wp[2][0];
            acc1[r] = bp1 + px * wp[0][1] + py * wp[1][1] + pz * wp[2][1];
        }
    }

    const int bk16 = half ? 16 : 0;   // B-fragment k-offset within a k-block

    // ---- two decoder blocks: net += c@Wc + bc ; net = leaky(net@Wb + bb) ----
#pragma unroll
    for (int blk = 0; blk < 2; ++blk) {
        const __bf16* wcT = blk ? s_wc2T : s_wc1T;
        const __bf16* wbT = blk ? s_wb2T : s_wb1T;
        const float*  bc  = blk ? b_c2 : b_c1;
        const float*  bb  = blk ? b_b2 : b_b1;

        // net += c @ Wc   (K = 128 in 4 blocks of 32, N = 32 in 2 tiles)
#pragma unroll
        for (int kb = 0; kb < 4; ++kb) {
            v16bf a  = load_frag32B(&s_c[wave][l15][kb * 32 + half * 16]);
            v16bf b0 = load_frag32B(&wcT[l15 * CPAD + kb * 32 + bk16]);
            v16bf b1 = load_frag32B(&wcT[(16 + l15) * CPAD + kb * 32 + bk16]);
            acc0 = __builtin_amdgcn_wmma_f32_16x16x32_bf16(false, a, false, b0,
                                                           (short)0, acc0, false, false);
            acc1 = __builtin_amdgcn_wmma_f32_16x16x32_bf16(false, a, false, b1,
                                                           (short)0, acc1, false, false);
        }
        {
            float bc0 = bc[l15], bc1 = bc[16 + l15];
#pragma unroll
            for (int r = 0; r < 8; ++r) { acc0[r] += bc0; acc1[r] += bc1; }
        }

        // spill net (bf16, column-swizzled) to LDS for re-read as an A matrix
        const int pos0 = swz32(l15), pos1 = swz32(16 + l15);
        __syncthreads();
#pragma unroll
        for (int r = 0; r < 8; ++r) {
            int m = half * 8 + r;
            s_net[wave][m][pos0] = (__bf16)acc0[r];
            s_net[wave][m][pos1] = (__bf16)acc1[r];
        }
        __syncthreads();

        // hidden = leaky_relu(net @ Wb + bb)   (single K=32 block)
        v8f h0, h1;
        {
            float bb0 = bb[l15], bb1 = bb[16 + l15];
#pragma unroll
            for (int r = 0; r < 8; ++r) { h0[r] = bb0; h1[r] = bb1; }
            v16bf a  = load_frag32B(&s_net[wave][l15][half * 16]);
            v16bf b0 = load_frag32B(&wbT[l15 * WBPAD + bk16]);
            v16bf b1 = load_frag32B(&wbT[(16 + l15) * WBPAD + bk16]);
            h0 = __builtin_amdgcn_wmma_f32_16x16x32_bf16(false, a, false, b0,
                                                         (short)0, h0, false, false);
            h1 = __builtin_amdgcn_wmma_f32_16x16x32_bf16(false, a, false, b1,
                                                         (short)0, h1, false, false);
        }
#pragma unroll
        for (int r = 0; r < 8; ++r) {
            acc0[r] = h0[r] > 0.0f ? h0[r] : h0[r] * NEG_SLOPE;
            acc1[r] = h1[r] > 0.0f ? h1[r] : h1[r] * NEG_SLOPE;
        }
    }

    // ---- out = net @ W_out + b_out : reduce 32 columns across 16 lanes ----
    {
        float w0 = W_out[l15], w1 = W_out[16 + l15];
        float bo = b_out[0];
        v8f part;
#pragma unroll
        for (int r = 0; r < 8; ++r) part[r] = acc0[r] * w0 + acc1[r] * w1;
#pragma unroll
        for (int m = 1; m < 16; m <<= 1) {
#pragma unroll
            for (int r = 0; r < 8; ++r) part[r] += __shfl_xor(part[r], m, 32);
        }
        if (l15 == 0) {
            size_t ob = (size_t)b * NPTS + tb + wave * PTS_PER_WAVE + half * 8;
#pragma unroll
            for (int r = 0; r < 8; ++r) out[ob + r] = part[r] + bo;
        }
    }
}

extern "C" void kernel_launch(void* const* d_in, const int* in_sizes, int n_in,
                              void* d_out, int out_size, void* d_ws, size_t ws_size,
                              hipStream_t stream) {
    (void)in_sizes; (void)n_in; (void)out_size;
    // d_in order: pcl, pcl_mem, c_plane, W_p, b_p, W_c1, b_c1, W_c2, b_c2,
    //             W_b1, b_b1, W_b2, b_b2, W_out, b_out
    const float* pcl_mem = (const float*)d_in[1];
    const float* c_plane = (const float*)d_in[2];
    const float* W_p  = (const float*)d_in[3];  const float* b_p  = (const float*)d_in[4];
    const float* W_c1 = (const float*)d_in[5];  const float* b_c1 = (const float*)d_in[6];
    const float* W_c2 = (const float*)d_in[7];  const float* b_c2 = (const float*)d_in[8];
    const float* W_b1 = (const float*)d_in[9];  const float* b_b1 = (const float*)d_in[10];
    const float* W_b2 = (const float*)d_in[11]; const float* b_b2 = (const float*)d_in[12];
    const float* W_out = (const float*)d_in[13]; const float* b_out = (const float*)d_in[14];
    float* out = (float*)d_out;

    dim3 grid(NB * (NPTS / PTS_PER_BLOCK));
    dim3 block(WAVES_PER_BLOCK * 32);

    const size_t vox_bytes = (size_t)NB * NG3 * NC * sizeof(__bf16);  // 256 MB
    if (ws_size >= vox_bytes) {
        __bf16* c_vox = (__bf16*)d_ws;
        transpose_kernel<<<dim3(NB * (NG3 / TVOX)), dim3(256), 0, stream>>>(c_plane, c_vox);
        local_decoder_kernel<true><<<grid, block, 0, stream>>>(
            pcl_mem, c_plane, c_vox, W_p, b_p, W_c1, b_c1, W_c2, b_c2,
            W_b1, b_b1, W_b2, b_b2, W_out, b_out, out);
    } else {
        local_decoder_kernel<false><<<grid, block, 0, stream>>>(
            pcl_mem, c_plane, (const __bf16*)nullptr, W_p, b_p, W_c1, b_c1, W_c2, b_c2,
            W_b1, b_b1, W_b2, b_b2, W_out, b_out, out);
    }
}
